// BetaModel_5660766896152
// MI455X (gfx1250) — matrix operand order repaired and checked by reading, not compiled, not run on hardware
//
#include <hip/hip_runtime.h>
#include <hip/hip_bf16.h>
#include <math.h>

typedef __attribute__((ext_vector_type(16))) _Float16 v16h;
typedef __attribute__((ext_vector_type(8)))  float    v8f;

// ---- model constants ----
constexpr int Dm = 280, Hn = 7, Ln = 7, FFn = 1120, Vn = 256, Tn = 512, Bn = 32, HDn = 40;
constexpr int BT = Bn * Tn;   // 16384 tokens
constexpr int DP = 288;       // D padded to 9*32
constexpr int BH = Bn * Hn;   // 224
constexpr int HDP = 64;       // head-dim padded (q/k, K-dim of QK^T)
constexpr int VDP = 48;       // head-dim padded (v, N-dim of PV)
constexpr int DPr = 320;      // weight rows padded to mult of 64 (D outputs)
constexpr int FFr = 1152;     // weight rows padded to mult of 64 (FF outputs)

#define DEV static __device__ __forceinline__

DEV uint4 ld16(const _Float16* p) { return *(const uint4*)(const void*)p; }

union ABu { uint4 u[2]; v16h v; };

DEV v8f wmma16(v16h a, v16h b, v8f c) {
  return __builtin_amdgcn_wmma_f32_16x16x32_f16(false, a, false, b, (short)0, c, false, false);
}

// one async 16B lane-copy global -> LDS (GV mode), tracked by ASYNCcnt
DEV void async_b128(const _Float16* g, _Float16* lds) {
  unsigned l = (unsigned)(uintptr_t)lds;  // low 32 bits of LDS aperture addr = wave-relative LDS offset
  asm volatile("global_load_async_to_lds_b128 %0, %1, off"
               :: "v"(l), "v"((unsigned long long)(uintptr_t)g)
               : "memory");
}
DEV void wait_async0() { asm volatile("s_wait_asynccnt 0x0" ::: "memory"); }

// -------------------- utility kernels --------------------
__global__ void zero_kernel(uint4* p, long n) {
  long g = (long)blockIdx.x * 256 + threadIdx.x;
  if (g < n) p[g] = make_uint4(0u, 0u, 0u, 0u);
}

// f32 [L,E,Din] -> f16 [L,Ep,Kp] zero-padded (rows AND cols)
__global__ void convpad_kernel(const float* __restrict__ src, _Float16* __restrict__ dst,
                               int L, int E, int Din, int Ep, int Kp) {
  long g = (long)blockIdx.x * 256 + threadIdx.x;
  long tot = (long)L * Ep * Kp;
  if (g >= tot) return;
  int kk = (int)(g % Kp);
  long r = g / Kp;
  int e = (int)(r % Ep);
  int l = (int)(r / Ep);
  float v = (e < E && kk < Din) ? src[((size_t)l * E + e) * Din + kk] : 0.f;
  dst[g] = (_Float16)v;
}

__global__ void embed_kernel(const int* __restrict__ idx, const float* __restrict__ ew,
                             float* __restrict__ x) {
  long g = (long)blockIdx.x * 256 + threadIdx.x;
  if (g >= (long)BT * Dm) return;
  int t = (int)(g / Dm), d = (int)(g % Dm);
  x[g] = ew[(size_t)idx[t] * Dm + d];
}

// wave-per-token RMSNorm: f32 x -> f16 h16 (padded cols stay zero)
__global__ void __launch_bounds__(256) rmsnorm_kernel(const float* __restrict__ x,
                                                      const float* __restrict__ w,
                                                      _Float16* __restrict__ h16) {
  const int lane = threadIdx.x & 31;
  const int tok = blockIdx.x * 8 + (threadIdx.x >> 5);
  if (tok >= BT) return;
  float v[9];
  float ss = 0.f;
#pragma unroll
  for (int i = 0; i < 9; i++) {
    int k = i * 32 + lane;
    v[i] = (k < Dm) ? x[(size_t)tok * Dm + k] : 0.f;
    ss += v[i] * v[i];
  }
#pragma unroll
  for (int m = 1; m < 32; m <<= 1) ss += __shfl_xor(ss, m, 32);
  float r = rsqrtf(ss / (float)Dm + 1e-6f);
#pragma unroll
  for (int i = 0; i < 9; i++) {
    int k = i * 32 + lane;
    if (k < Dm) h16[(size_t)tok * DP + k] = (_Float16)(v[i] * r * w[k]);
  }
}

// in-place RoPE on f16 q/k in [bh][t][HDP] layout; one thread per (bh,t,pair)
__global__ void rope_kernel(_Float16* __restrict__ q, _Float16* __restrict__ k) {
  long g = (long)blockIdx.x * 256 + threadIdx.x;
  if (g >= (long)BH * Tn * (HDn / 2)) return;
  int p = (int)(g % 20);
  long r = g / 20;
  int t = (int)(r % Tn);
  int bh = (int)(r / Tn);
  float ang = (float)t * __expf(-__logf(10000.f) * (float)p / 20.f);
  float c = cosf(ang), s = sinf(ang);
  size_t base = ((size_t)bh * Tn + t) * HDP;
  float q0 = (float)q[base + p], q1 = (float)q[base + p + 20];
  q[base + p]      = (_Float16)(q0 * c - q1 * s);
  q[base + p + 20] = (_Float16)(q1 * c + q0 * s);
  float k0 = (float)k[base + p], k1 = (float)k[base + p + 20];
  k[base + p]      = (_Float16)(k0 * c - k1 * s);
  k[base + p + 20] = (_Float16)(k1 * c + k0 * s);
}

// -------------------- panel GEMM with async-LDS B staging --------------------
// C[M=BT, N] = A[M,Kp] * B[N,Kp]^T. Workgroup tile = 128(M) x 64(N); wave w
// owns rows w*16..w*16+15 and all 4 N-subtiles. The 64x32 f16 B panel is
// double-buffered in LDS via global_load_async_to_lds_b128 (ASYNCcnt).
// EPI: 0 = store f32, 1 = += f32 residual, 2 = q/k head-layout f16 scatter,
//      3 = v transposed f16 scatter, 4 = dual-B fused SiLU-gate (f16 store)
template <int EPI>
__global__ void __launch_bounds__(256) gemm_kernel(const _Float16* __restrict__ A,
                                                   const _Float16* __restrict__ B,
                                                   const _Float16* __restrict__ B2,
                                                   int Kp, int tilesN, int Nvalid,
                                                   float* __restrict__ Cf, int ldc,
                                                   _Float16* __restrict__ Ch) {
  constexpr bool DUAL = (EPI == 4);
  constexpr int NB2 = DUAL ? 2 : 1;
  alignas(16) __shared__ _Float16 pB[2][64][32];
  alignas(16) __shared__ _Float16 pB2[NB2][64][32];

  const int tid = threadIdx.x;
  const int lane = tid & 31, w = tid >> 5;
  const int nGroups = (tilesN + 3) >> 2;
  const int mBlk = blockIdx.x / nGroups;
  const int ng = blockIdx.x % nGroups;
  const int n16 = lane & 15, sel = lane >> 4;
  const int groupRow = ng * 64;

  const _Float16* pa = A + (size_t)(mBlk * 128 + w * 16 + n16) * Kp;

  // per-thread async copy slot: thread -> (row, 16B part) of the 64x32 panel
  const int lrow = tid >> 2;
  const int lpart = (tid & 3) * 8;
  const _Float16* gb = B + (size_t)(groupRow + lrow) * Kp + lpart;
  const _Float16* gb2 = DUAL ? (B2 + (size_t)(groupRow + lrow) * Kp + lpart) : nullptr;

  v8f c[4] = {};
  v8f c2[4] = {};

  const int nK = Kp / 32;
  // prime buffer 0
  async_b128(gb, &pB[0][lrow][lpart]);
  if constexpr (DUAL) async_b128(gb2, &pB2[0][lrow][lpart]);

  for (int kc = 0; kc < nK; kc++) {
    wait_async0();        // my wave's panel chunk arrived
    __syncthreads();      // everyone's chunk arrived; prev buffer fully read
    const int cur = kc & 1;
    if (kc + 1 < nK) {
      const int nxt = (kc + 1) & 1;
      async_b128(gb + (kc + 1) * 32, &pB[nxt][lrow][lpart]);
      if constexpr (DUAL) async_b128(gb2 + (kc + 1) * 32, &pB2[nxt][lrow][lpart]);
    }
    ABu a;
    a.u[0] = ld16(pa + kc * 32 + sel * 8);
    a.u[1] = ld16(pa + kc * 32 + 16 + sel * 8);
#pragma unroll
    for (int tn = 0; tn < 4; tn++) {
      ABu b;
      b.u[0] = ld16(&pB[cur][tn * 16 + n16][sel * 8]);
      b.u[1] = ld16(&pB[cur][tn * 16 + n16][16 + sel * 8]);
      c[tn] = wmma16(a.v, b.v, c[tn]);
      if constexpr (DUAL) {
        ABu d;
        d.u[0] = ld16(&pB2[cur][tn * 16 + n16][sel * 8]);
        d.u[1] = ld16(&pB2[cur][tn * 16 + n16][16 + sel * 8]);
        c2[tn] = wmma16(a.v, d.v, c2[tn]);
      }
    }
  }

#pragma unroll
  for (int tn = 0; tn < 4; tn++) {
    int n = groupRow + tn * 16 + n16;
#pragma unroll
    for (int i = 0; i < 8; i++) {
      int m = mBlk * 128 + w * 16 + i + sel * 8;
      float v = c[tn][i];
      if constexpr (EPI == 0) {
        if (n < Nvalid) Cf[(size_t)m * ldc + n] = v;
      } else if constexpr (EPI == 1) {
        if (n < Nvalid) Cf[(size_t)m * ldc + n] += v;
      } else if constexpr (EPI == 2) {
        if (n < Dm) {
          int h = n / HDn, hd = n % HDn, bb = m >> 9, t = m & (Tn - 1);
          Ch[(((size_t)(bb * Hn + h)) * Tn + t) * HDP + hd] = (_Float16)v;
        }
      } else if constexpr (EPI == 3) {
        if (n < Dm) {
          int h = n / HDn, hd = n % HDn, bb = m >> 9, t = m & (Tn - 1);
          Ch[(((size_t)(bb * Hn + h)) * VDP + hd) * Tn + t] = (_Float16)v;
        }
      } else {
        if (n < Nvalid) {
          float sg = v / (1.f + __expf(-v));
          Ch[(size_t)m * FFn + n] = (_Float16)(sg * c2[tn][i]);
        }
      }
    }
  }
}

// -------------------- flash attention (wave per 16-query tile) --------------------
__global__ void __launch_bounds__(256) flash_kernel(const _Float16* __restrict__ qf,
                                                    const _Float16* __restrict__ kf,
                                                    const _Float16* __restrict__ vT,
                                                    _Float16* __restrict__ o16) {
  alignas(16) __shared__ _Float16 psh[8][16][32];
  const int lane = threadIdx.x & 31;
  const int w = threadIdx.x >> 5;
  const int wid = blockIdx.x * 8 + w;
  const int bh = wid >> 5;
  const int qt = wid & 31;
  if (bh >= BH) return;
  const int qbase = qt * 16;
  const int n16 = lane & 15, sel = lane >> 4;

  const _Float16* qrow = qf + ((size_t)bh * Tn + qbase + n16) * HDP;
  ABu qa0, qa1;
  qa0.u[0] = ld16(qrow + sel * 8);
  qa0.u[1] = ld16(qrow + 16 + sel * 8);
  qa1.u[0] = ld16(qrow + 32 + sel * 8);
  qa1.u[1] = ld16(qrow + 48 + sel * 8);

  v8f zero = {};
  v8f oacc[3] = {zero, zero, zero};
  float rmax[8], rsum[8];
#pragma unroll
  for (int i = 0; i < 8; i++) { rmax[i] = -INFINITY; rsum[i] = 0.f; }

  const float scale = 0.15811388300841897f;  // 1/sqrt(40)

  for (int jb = 0; jb < qbase + 16; jb += 32) {
    ABu kb;
    v8f s0 = {}, s1 = {};
    const _Float16* kr0 = kf + ((size_t)bh * Tn + jb + n16) * HDP + sel * 8;
    const _Float16* kr1 = kf + ((size_t)bh * Tn + jb + 16 + n16) * HDP + sel * 8;
    kb.u[0] = ld16(kr0);      kb.u[1] = ld16(kr0 + 16); s0 = wmma16(qa0.v, kb.v, s0);
    kb.u[0] = ld16(kr0 + 32); kb.u[1] = ld16(kr0 + 48); s0 = wmma16(qa1.v, kb.v, s0);
    kb.u[0] = ld16(kr1);      kb.u[1] = ld16(kr1 + 16); s1 = wmma16(qa0.v, kb.v, s1);
    kb.u[0] = ld16(kr1 + 32); kb.u[1] = ld16(kr1 + 48); s1 = wmma16(qa1.v, kb.v, s1);

#pragma unroll
    for (int i = 0; i < 8; i++) {
      int qm = qbase + i + sel * 8;
      float a0 = (jb + n16 <= qm) ? s0[i] * scale : -INFINITY;
      float a1 = (jb + 16 + n16 <= qm) ? s1[i] * scale : -INFINITY;
      float mx = fmaxf(a0, a1);
      mx = fmaxf(mx, __shfl_xor(mx, 1, 32));
      mx = fmaxf(mx, __shfl_xor(mx, 2, 32));
      mx = fmaxf(mx, __shfl_xor(mx, 4, 32));
      mx = fmaxf(mx, __shfl_xor(mx, 8, 32));
      float nm = fmaxf(rmax[i], mx);
      float alpha = __expf(rmax[i] - nm);
      float p0 = __expf(a0 - nm), p1 = __expf(a1 - nm);
      float ps = p0 + p1;
      ps += __shfl_xor(ps, 1, 32);
      ps += __shfl_xor(ps, 2, 32);
      ps += __shfl_xor(ps, 4, 32);
      ps += __shfl_xor(ps, 8, 32);
      rsum[i] = rsum[i] * alpha + ps;
      rmax[i] = nm;
#pragma unroll
      for (int t3 = 0; t3 < 3; t3++) oacc[t3][i] *= alpha;
      psh[w][i + sel * 8][n16] = (_Float16)p0;
      psh[w][i + sel * 8][n16 + 16] = (_Float16)p1;
    }
    asm volatile("s_wait_dscnt 0x0" ::: "memory");

    ABu pa;
    pa.u[0] = ld16(&psh[w][n16][sel * 8]);
    pa.u[1] = ld16(&psh[w][n16][16 + sel * 8]);

#pragma unroll
    for (int dt = 0; dt < 3; dt++) {
      const _Float16* vr = vT + ((size_t)bh * VDP + dt * 16 + n16) * Tn + jb + sel * 8;
      ABu vb;
      vb.u[0] = ld16(vr);
      vb.u[1] = ld16(vr + 16);
      oacc[dt] = wmma16(pa.v, vb.v, oacc[dt]);
    }
  }

  const int b = bh / Hn, h = bh % Hn;
#pragma unroll
  for (int i = 0; i < 8; i++) {
    float inv = 1.f / rsum[i];
    int tok = b * Tn + qbase + i + sel * 8;
#pragma unroll
    for (int dt = 0; dt < 3; dt++) {
      int d = dt * 16 + n16;
      if (d < HDn)
        o16[(size_t)tok * DP + h * HDn + d] = (_Float16)(oacc[dt][i] * inv);
    }
  }
}

// -------------------- host orchestration --------------------
extern "C" void kernel_launch(void* const* d_in, const int* in_sizes, int n_in,
                              void* d_out, int out_size, void* d_ws, size_t ws_size,
                              hipStream_t stream) {
  (void)in_sizes; (void)n_in; (void)out_size; (void)ws_size;
  const int*   idx = (const int*)d_in[0];
  const float* ew  = (const float*)d_in[1];
  const float* wq  = (const float*)d_in[2];
  const float* wk  = (const float*)d_in[3];
  const float* wv  = (const float*)d_in[4];
  const float* wo  = (const float*)d_in[5];
  const float* w1  = (const float*)d_in[6];
  const float* w2  = (const float*)d_in[7];
  const float* w3  = (const float*)d_in[8];
  const float* n1  = (const float*)d_in[9];
  const float* n2  = (const float*)d_in[10];
  const float* nw  = (const float*)d_in[11];
  float* out = (float*)d_out;

  char* p = (char*)d_ws;
  auto alloc = [&](size_t bytes) -> char* {
    char* r = p;
    p += (bytes + 255) & ~(size_t)255;
    return r;
  };

  // zero-init region first (contiguous: sizes are multiples of 256B)
  _Float16* h16 = (_Float16*)alloc((size_t)BT * DP * 2);
  _Float16* o16 = (_Float16*)alloc((size_t)BT * DP * 2);
  _Float16* qf  = (_Float16*)alloc((size_t)BH * Tn * HDP * 2);
  _Float16* kf  = (_Float16*)alloc((size_t)BH * Tn * HDP * 2);
  _Float16* vT  = (_Float16*)alloc((size_t)BH * VDP * Tn * 2);
  size_t zeroBytes = (size_t)(p - (char*)h16);
  float*    x    = (float*)alloc((size_t)BT * Dm * 4);
  _Float16* g16  = (_Float16*)alloc((size_t)BT * FFn * 2);
  _Float16* wq16 = (_Float16*)alloc((size_t)Ln * DPr * DP * 2);
  _Float16* wk16 = (_Float16*)alloc((size_t)Ln * DPr * DP * 2);
  _Float16* wv16 = (_Float16*)alloc((size_t)Ln * DPr * DP * 2);
  _Float16* wo16 = (_Float16*)alloc((size_t)Ln * DPr * DP * 2);
  _Float16* w116 = (_Float16*)alloc((size_t)Ln * FFr * DP * 2);
  _Float16* w316 = (_Float16*)alloc((size_t)Ln * FFr * DP * 2);
  _Float16* w216 = (_Float16*)alloc((size_t)Ln * DPr * FFn * 2);
  _Float16* e16  = (_Float16*)alloc((size_t)Vn * DP * 2);

  auto g1 = [](long n) { return dim3((unsigned)((n + 255) / 256)); };
  auto gemmGrid = [](int tilesN) {
    return dim3((unsigned)((BT / 128) * ((tilesN + 3) >> 2)));
  };

  long nz = (long)(zeroBytes / 16);
  zero_kernel<<<g1(nz), 256, 0, stream>>>((uint4*)h16, nz);

  convpad_kernel<<<g1((long)Ln * DPr * DP), 256, 0, stream>>>(wq, wq16, Ln, Dm, Dm, DPr, DP);
  convpad_kernel<<<g1((long)Ln * DPr * DP), 256, 0, stream>>>(wk, wk16, Ln, Dm, Dm, DPr, DP);
  convpad_kernel<<<g1((long)Ln * DPr * DP), 256, 0, stream>>>(wv, wv16, Ln, Dm, Dm, DPr, DP);
  convpad_kernel<<<g1((long)Ln * DPr * DP), 256, 0, stream>>>(wo, wo16, Ln, Dm, Dm, DPr, DP);
  convpad_kernel<<<g1((long)Ln * FFr * DP), 256, 0, stream>>>(w1, w116, Ln, FFn, Dm, FFr, DP);
  convpad_kernel<<<g1((long)Ln * FFr * DP), 256, 0, stream>>>(w3, w316, Ln, FFn, Dm, FFr, DP);
  convpad_kernel<<<g1((long)Ln * DPr * FFn), 256, 0, stream>>>(w2, w216, Ln, Dm, FFn, DPr, FFn);
  convpad_kernel<<<g1((long)Vn * DP), 256, 0, stream>>>(ew, e16, 1, Vn, Dm, Vn, DP);

  embed_kernel<<<g1((long)BT * Dm), 256, 0, stream>>>(idx, ew, x);

  const size_t wOffD = (size_t)DPr * DP;
  const size_t wOffF = (size_t)FFr * DP;
  const size_t wOff2 = (size_t)DPr * FFn;
  for (int l = 0; l < Ln; l++) {
    rmsnorm_kernel<<<BT / 8, 256, 0, stream>>>(x, n1 + (size_t)l * Dm, h16);
    gemm_kernel<2><<<gemmGrid(18), 256, 0, stream>>>(h16, wq16 + l * wOffD, nullptr, DP, 18, Dm, nullptr, 0, qf);
    gemm_kernel<2><<<gemmGrid(18), 256, 0, stream>>>(h16, wk16 + l * wOffD, nullptr, DP, 18, Dm, nullptr, 0, kf);
    gemm_kernel<3><<<gemmGrid(18), 256, 0, stream>>>(h16, wv16 + l * wOffD, nullptr, DP, 18, Dm, nullptr, 0, vT);
    rope_kernel<<<g1((long)BH * Tn * (HDn / 2)), 256, 0, stream>>>(qf, kf);
    flash_kernel<<<(BH * 32 + 7) / 8, 256, 0, stream>>>(qf, kf, vT, o16);
    gemm_kernel<1><<<gemmGrid(18), 256, 0, stream>>>(o16, wo16 + l * wOffD, nullptr, DP, 18, Dm, x, Dm, nullptr);
    rmsnorm_kernel<<<BT / 8, 256, 0, stream>>>(x, n2 + (size_t)l * Dm, h16);
    gemm_kernel<4><<<gemmGrid(70), 256, 0, stream>>>(h16, w116 + l * wOffF, w316 + l * wOffF, DP, 70, FFn, nullptr, 0, g16);
    gemm_kernel<1><<<gemmGrid(18), 256, 0, stream>>>(g16, w216 + l * wOff2, nullptr, FFn, 18, Dm, x, Dm, nullptr);
  }

  rmsnorm_kernel<<<BT / 8, 256, 0, stream>>>(x, nw, h16);
  gemm_kernel<0><<<gemmGrid(16), 256, 0, stream>>>(h16, e16, nullptr, DP, 16, Vn, out, Vn, nullptr);
}
